// SAGEModel_34797825032691
// MI455X (gfx1250) — compile-verified, matrix operand order checked by reading, not run on hardware
//
#include <hip/hip_runtime.h>
#include <hip/hip_bf16.h>

typedef __attribute__((ext_vector_type(16))) _Float16 v16h;
typedef __attribute__((ext_vector_type(8)))  float    v8f;

union V16 { v16h v; uint4 u[2]; };

#define D 128
#define NNODES 10000
#define NEDGES 640000

// ---------------- utility kernels ----------------

__global__ void zero_f32(float* __restrict__ p, int n) {
    int i = blockIdx.x * blockDim.x + threadIdx.x;
    if (i < n) p[i] = 0.0f;
}

__global__ void deg_kernel(const int* __restrict__ dst, float* __restrict__ deg, int nE) {
    int e = blockIdx.x * blockDim.x + threadIdx.x;
    if (e < nE) atomicAdd(&deg[dst[e]], 1.0f);
}

__global__ void cvt_f16(const float* __restrict__ in, _Float16* __restrict__ out, int n) {
    int i = blockIdx.x * blockDim.x + threadIdx.x;
    if (i < n) out[i] = (_Float16)in[i];
}

// W [128][128] row-major f32 -> Wt [n][k] f16 (column of W becomes contiguous row)
__global__ void transpose128(const float* __restrict__ W, _Float16* __restrict__ Wt) {
    int i = blockIdx.x * blockDim.x + threadIdx.x;   // 16384
    if (i < D * D) {
        int n = i >> 7, k = i & 127;
        Wt[n * D + k] = (_Float16)W[k * D + n];
    }
}

// W_pred [256][2] f32 -> WpT [16][256] f16, cols 2..15 zero-padded
__global__ void transpose_pred(const float* __restrict__ W, _Float16* __restrict__ Wt) {
    int i = blockIdx.x * blockDim.x + threadIdx.x;   // 4096
    if (i < 16 * 256) {
        int n = i >> 8, k = i & 255;
        float v = (n < 2) ? W[k * 2 + n] : 0.0f;
        Wt[n * 256 + k] = (_Float16)v;
    }
}

// segment-sum scatter, f32 source: agg[dst] += x[src], 4 feats/thread
__global__ void scatter_f32(const float* __restrict__ xsrc, const int* __restrict__ src,
                            const int* __restrict__ dst, float* __restrict__ agg, int nE) {
    int t = blockIdx.x * blockDim.x + threadIdx.x;   // nE * 32
    int e = t >> 5;
    if (e >= nE) return;
    int c = (t & 31) << 2;
    int s = src[e], d = dst[e];
    const float4 v = *(const float4*)(xsrc + s * D + c);
    atomicAdd(&agg[d * D + c + 0], v.x);
    atomicAdd(&agg[d * D + c + 1], v.y);
    atomicAdd(&agg[d * D + c + 2], v.z);
    atomicAdd(&agg[d * D + c + 3], v.w);
}

// segment-sum scatter, f16 source
__global__ void scatter_f16(const _Float16* __restrict__ hsrc, const int* __restrict__ src,
                            const int* __restrict__ dst, float* __restrict__ agg, int nE) {
    int t = blockIdx.x * blockDim.x + threadIdx.x;
    int e = t >> 5;
    if (e >= nE) return;
    int c = (t & 31) << 2;
    int s = src[e], d = dst[e];
    const _Float16* p = hsrc + s * D + c;
    atomicAdd(&agg[d * D + c + 0], (float)p[0]);
    atomicAdd(&agg[d * D + c + 1], (float)p[1]);
    atomicAdd(&agg[d * D + c + 2], (float)p[2]);
    atomicAdd(&agg[d * D + c + 3], (float)p[3]);
}

// aggh = f16( agg / max(deg,1) )
__global__ void divcvt(const float* __restrict__ agg, const float* __restrict__ deg,
                       _Float16* __restrict__ aggh, int n) {
    int i = blockIdx.x * blockDim.x + threadIdx.x;
    if (i < n) {
        int node = i >> 7;
        float dg = fmaxf(deg[node], 1.0f);
        aggh[i] = (_Float16)(agg[i] / dg);
    }
}

// ---------------- WMMA GEMM: h = act(A@WsT^T + G@WnT^T + b) ----------------
// Grid: 625 blocks (16 rows each), 256 threads = 8 waves, wave w owns N-cols [16w,16w+16)
__global__ void sage_layer_wmma(const _Float16* __restrict__ Ah,
                                const _Float16* __restrict__ Gh,
                                const _Float16* __restrict__ WsT,  // [n][k] f16
                                const _Float16* __restrict__ WnT,  // [n][k] f16
                                const float*    __restrict__ bias, // [128] f32
                                _Float16*       __restrict__ outH,
                                int relu) {
    __shared__ __align__(16) _Float16 sA[16 * D];
    __shared__ __align__(16) _Float16 sG[16 * D];
    const int tid   = threadIdx.x;
    const int mbase = blockIdx.x * 16;
    {   // cooperative 128-bit staged loads: 256 thr x 8 halfs = one 16x128 tile each
        int r = tid >> 4;
        int c = (tid & 15) << 3;
        *(uint4*)(sA + r * D + c) = *(const uint4*)(Ah + (mbase + r) * D + c);
        *(uint4*)(sG + r * D + c) = *(const uint4*)(Gh + (mbase + r) * D + c);
    }
    __syncthreads();

    const int wave = tid >> 5;
    const int lane = tid & 31;
    const int lr   = lane & 15;       // A row / B,C col within tile
    const int hi   = lane >> 4;       // lane half selects K sub-chunks
    const int nbase = wave * 16;

    v8f c = {};
#pragma unroll
    for (int kk = 0; kk < D; kk += 32) {
        const int k0 = kk + hi * 8;
        V16 a, g, bs, bn;
        a.u[0]  = *(const uint4*)(sA + lr * D + k0);
        a.u[1]  = *(const uint4*)(sA + lr * D + k0 + 16);
        g.u[0]  = *(const uint4*)(sG + lr * D + k0);
        g.u[1]  = *(const uint4*)(sG + lr * D + k0 + 16);
        const _Float16* ws = WsT + (nbase + lr) * D;
        bs.u[0] = *(const uint4*)(ws + k0);
        bs.u[1] = *(const uint4*)(ws + k0 + 16);
        const _Float16* wn = WnT + (nbase + lr) * D;
        bn.u[0] = *(const uint4*)(wn + k0);
        bn.u[1] = *(const uint4*)(wn + k0 + 16);
        c = __builtin_amdgcn_wmma_f32_16x16x32_f16(false, a.v, false, bs.v, (short)0, c, false, false);
        c = __builtin_amdgcn_wmma_f32_16x16x32_f16(false, g.v, false, bn.v, (short)0, c, false, false);
    }

    const int col = nbase + lr;
    const float b = bias[col];
#pragma unroll
    for (int v = 0; v < 8; ++v) {
        float val = c[v] + b;
        if (relu) val = fmaxf(val, 0.0f);
        int row = mbase + v + hi * 8;
        outH[row * D + col] = (_Float16)val;
    }
}

// ---------------- Edge predictor: concat(h[src],h[dst]) @ W_pred + b ----------------
// 16 edges per wave, K=256 (first 128 from src row, next 128 from dst row)
__global__ void edge_pred_wmma(const _Float16* __restrict__ Hh,
                               const int* __restrict__ src,
                               const int* __restrict__ dst,
                               const _Float16* __restrict__ WpT,   // [16][256] f16
                               const float* __restrict__ bpred,    // [2] f32
                               float* __restrict__ out) {          // [E][2] f32
    const int tid  = threadIdx.x;
    const int wave = tid >> 5;
    const int lane = tid & 31;
    const int lr   = lane & 15;
    const int hi   = lane >> 4;
    const int tile = blockIdx.x * 8 + wave;
    const int ebase = tile * 16;
    const int e = ebase + lr;
    const int srow = src[e] * D;
    const int drow = dst[e] * D;

    v8f c = {};
#pragma unroll
    for (int kk = 0; kk < 2 * D; kk += 32) {
        const int k0 = kk + hi * 8;
        const int k1 = k0 + 16;
        // chunk base is 8-aligned, never straddles the 128-col concat boundary
        const _Float16* p0 = Hh + ((k0 < D) ? (srow + k0) : (drow + k0 - D));
        const _Float16* p1 = Hh + ((k1 < D) ? (srow + k1) : (drow + k1 - D));
        V16 a, b;
        a.u[0] = *(const uint4*)p0;
        a.u[1] = *(const uint4*)p1;
        const _Float16* wp = WpT + lr * 256;
        b.u[0] = *(const uint4*)(wp + k0);
        b.u[1] = *(const uint4*)(wp + k1);
        c = __builtin_amdgcn_wmma_f32_16x16x32_f16(false, a.v, false, b.v, (short)0, c, false, false);
    }

    if (lr < 2) {
        const float bb = bpred[lr];
#pragma unroll
        for (int v = 0; v < 8; ++v) {
            int row = ebase + v + hi * 8;
            out[row * 2 + lr] = c[v] + bb;
        }
    }
}

// ---------------- host-side orchestration ----------------

extern "C" void kernel_launch(void* const* d_in, const int* in_sizes, int n_in,
                              void* d_out, int out_size, void* d_ws, size_t ws_size,
                              hipStream_t stream) {
    const float* x       = (const float*)d_in[0];
    const int*   src     = (const int*)  d_in[1];
    const int*   dst     = (const int*)  d_in[2];
    const float* W_self1 = (const float*)d_in[3];
    const float* W_neigh1= (const float*)d_in[4];
    const float* b1      = (const float*)d_in[5];
    const float* W_self2 = (const float*)d_in[6];
    const float* W_neigh2= (const float*)d_in[7];
    const float* b2      = (const float*)d_in[8];
    const float* W_pred  = (const float*)d_in[9];
    const float* b_pred  = (const float*)d_in[10];
    float* out = (float*)d_out;

    // carve workspace (~15.6 MB total)
    char* ws = (char*)d_ws;
    size_t off = 0;
    auto carve = [&](size_t bytes) {
        char* p = ws + off;
        off += (bytes + 255) & ~(size_t)255;
        return p;
    };
    float*    deg  = (float*)   carve(NNODES * sizeof(float));
    _Float16* xh   = (_Float16*)carve((size_t)NNODES * D * 2);
    _Float16* ws1t = (_Float16*)carve(D * D * 2);
    _Float16* wn1t = (_Float16*)carve(D * D * 2);
    _Float16* ws2t = (_Float16*)carve(D * D * 2);
    _Float16* wn2t = (_Float16*)carve(D * D * 2);
    _Float16* wpt  = (_Float16*)carve(16 * 256 * 2);
    float*    agg  = (float*)   carve((size_t)NNODES * D * 4);
    _Float16* aggh = (_Float16*)carve((size_t)NNODES * D * 2);
    _Float16* h1h  = (_Float16*)carve((size_t)NNODES * D * 2);
    _Float16* h2h  = (_Float16*)carve((size_t)NNODES * D * 2);
    (void)ws_size; (void)n_in; (void)in_sizes; (void)out_size;

    const int NF = NNODES * D;            // 1,280,000
    const int T  = 256;

    // degree (shared by both layers)
    zero_f32<<<(NNODES + T - 1) / T, T, 0, stream>>>(deg, NNODES);
    deg_kernel<<<(NEDGES + T - 1) / T, T, 0, stream>>>(dst, deg, NEDGES);

    // precision conversions / weight transposes
    cvt_f16<<<(NF + T - 1) / T, T, 0, stream>>>(x, xh, NF);
    transpose128<<<(D * D + T - 1) / T, T, 0, stream>>>(W_self1,  ws1t);
    transpose128<<<(D * D + T - 1) / T, T, 0, stream>>>(W_neigh1, wn1t);
    transpose128<<<(D * D + T - 1) / T, T, 0, stream>>>(W_self2,  ws2t);
    transpose128<<<(D * D + T - 1) / T, T, 0, stream>>>(W_neigh2, wn2t);
    transpose_pred<<<(16 * 256 + T - 1) / T, T, 0, stream>>>(W_pred, wpt);

    // layer 1: agg = mean(x[src] -> dst); h1 = relu(x@Ws1 + agg@Wn1 + b1)
    zero_f32<<<(NF + T - 1) / T, T, 0, stream>>>(agg, NF);
    scatter_f32<<<(NEDGES * 32) / T, T, 0, stream>>>(x, src, dst, agg, NEDGES);
    divcvt<<<(NF + T - 1) / T, T, 0, stream>>>(agg, deg, aggh, NF);
    sage_layer_wmma<<<NNODES / 16, T, 0, stream>>>(xh, aggh, ws1t, wn1t, b1, h1h, 1);

    // layer 2: agg = mean(h1[src] -> dst); h2 = h1@Ws2 + agg@Wn2 + b2
    zero_f32<<<(NF + T - 1) / T, T, 0, stream>>>(agg, NF);
    scatter_f16<<<(NEDGES * 32) / T, T, 0, stream>>>(h1h, src, dst, agg, NEDGES);
    divcvt<<<(NF + T - 1) / T, T, 0, stream>>>(agg, deg, aggh, NF);
    sage_layer_wmma<<<NNODES / 16, T, 0, stream>>>(h1h, aggh, ws2t, wn2t, b2, h2h, 0);

    // edge predictor: 640000 edges / 16 per wave / 8 waves per block = 5000 blocks
    edge_pred_wmma<<<NEDGES / (16 * 8), T, 0, stream>>>(h2h, src, dst, wpt, b_pred, out);
}
